// DsPillarEncoder2xHA_8942121910508
// MI455X (gfx1250) — matrix operand order repaired and checked by reading, not compiled, not run on hardware
//
#include <hip/hip_runtime.h>

typedef __bf16 bf16_t;
typedef __attribute__((ext_vector_type(16))) __bf16 v16bf;
typedef __attribute__((ext_vector_type(8)))  __bf16 v8bf;
typedef __attribute__((ext_vector_type(8)))  float  v8f;

#define BEV_RES 0.2f
#define XY_MIN  (-51.2f)
#define GRID_BEV 512
#define EPS_BN  1e-3f

// ---------------------------------------------------------------------------
// Pillar scatter-max: per point, build 8-dim feature, 8->64 MLP + BN + ReLU,
// atomicMax into fp32 NHWC BEV grid [512][512][64]. Non-negative floats
// compare correctly as unsigned ints, and the grid is zero-initialized, so
// uint atomicMax is exact.
// ---------------------------------------------------------------------------
__global__ __launch_bounds__(256)
void pillar_scatter(const float* __restrict__ xyz, const float* __restrict__ pfeat,
                    const float* __restrict__ mlpw,
                    const float* __restrict__ g, const float* __restrict__ b,
                    const float* __restrict__ m, const float* __restrict__ v,
                    float* __restrict__ bev, int N)
{
    __shared__ float sW[8 * 64];
    __shared__ float sS[64];   // BN scale
    __shared__ float sT[64];   // BN shift
    int t = threadIdx.x;
    for (int i = t; i < 8 * 64; i += blockDim.x) sW[i] = mlpw[i];
    if (t < 64) {
        float sc = g[t] * rsqrtf(v[t] + EPS_BN);
        sS[t] = sc;
        sT[t] = b[t] - m[t] * sc;
    }
    __syncthreads();

    int idx = blockIdx.x * blockDim.x + t;
    if (idx >= N) return;

    float x = xyz[idx * 3 + 0];
    float y = xyz[idx * 3 + 1];
    float z = xyz[idx * 3 + 2];
    int ix = (int)floorf((x - XY_MIN) / BEV_RES);
    int iy = (int)floorf((y - XY_MIN) / BEV_RES);
    if (ix < 0 || ix >= GRID_BEV || iy < 0 || iy >= GRID_BEV) return;

    float cx = XY_MIN + ((float)ix + 0.5f) * BEV_RES;
    float cy = XY_MIN + ((float)iy + 0.5f) * BEV_RES;
    float f[8] = { x, y, z, x - cx, y - cy, z,
                   pfeat[idx * 2 + 0], pfeat[idx * 2 + 1] };

    unsigned int* cell = (unsigned int*)(bev + ((size_t)iy * GRID_BEV + ix) * 64);
    for (int c = 0; c < 64; ++c) {
        float h = 0.f;
        #pragma unroll
        for (int k = 0; k < 8; ++k) h += f[k] * sW[k * 64 + c];
        h = h * sS[c] + sT[c];
        if (h > 0.f) atomicMax(cell + c, __float_as_uint(h));
    }
}

// ---------------------------------------------------------------------------
// fp32 -> bf16 elementwise conversion
// ---------------------------------------------------------------------------
__global__ __launch_bounds__(256)
void f32_to_bf16(const float* __restrict__ in, bf16_t* __restrict__ out, int n)
{
    int i = blockIdx.x * blockDim.x + threadIdx.x;
    if (i < n) out[i] = (bf16_t)in[i];
}

// ---------------------------------------------------------------------------
// Weight prep: fold BN scale into weights and swizzle into WMMA B-fragment
// layout for v_wmma_f32_16x16x32_bf16:
//   B is 32x16 (KxN) bf16, 8 VGPRs/lane: lane l holds column n=l%16,
//   K half = l/16, element e (0..15) of the v16bf holds k = (l/16)*16 + e (contiguous order chosen here; A loads match).
// Flat layout: wf[((kp*(CI/32)+cc)*(CO/16)+ct)*512 + l*16 + e]
// Also emits fused bias (beta - mean*scale).
// ---------------------------------------------------------------------------
__global__ __launch_bounds__(256)
void prep_weights(const float* __restrict__ w, const float* __restrict__ g,
                  const float* __restrict__ b, const float* __restrict__ m,
                  const float* __restrict__ v,
                  bf16_t* __restrict__ wf, float* __restrict__ bias,
                  int CI, int CO)
{
    int t = blockIdx.x * blockDim.x + threadIdx.x;
    if (t < CO) {
        float sc = g[t] * rsqrtf(v[t] + EPS_BN);
        bias[t] = b[t] - m[t] * sc;
    }
    int total = 9 * CI * CO;
    if (t >= total) return;

    int nCT = CO >> 4;
    int nCC = CI >> 5;
    int e  = t & 15;
    int l  = (t >> 4) & 31;
    int q  = t >> 9;
    int ct = q % nCT;
    int cc = (q / nCT) % nCC;
    int kp = q / (nCT * nCC);

    int n  = l & 15;
    int kh = l >> 4;
    int co = ct * 16 + n;
    int ci = cc * 32 + kh * 16 + e;
    int ky = kp / 3, kx = kp % 3;

    float sc  = g[co] * rsqrtf(v[co] + EPS_BN);
    float val = w[((size_t)(co * CI + ci) * 3 + ky) * 3 + kx] * sc;
    wf[t] = (bf16_t)val;
}

// ---------------------------------------------------------------------------
// Implicit-GEMM 3x3 conv, bf16 WMMA with fp32 accumulate.
// All shape parameters are compile-time: div/mod become shifts, the channel
// loop fully unrolls into a dense WMMA stream.
// One wave -> 16 output pixels x 64 output channels (4 accumulators): each
// gathered A-fragment (2x16B bounds-checked loads, concatenated with a
// single shufflevector so the halves land directly in the WMMA source
// registers) is reused by 4 WMMAs whose B-fragments are single contiguous
// 32B loads from pre-swizzled weights. 8 waves/block = 128 pixels.
// Epilogue fuses bias + ReLU, stores bf16 NHWC (+ optional fp32 NCHW).
// ---------------------------------------------------------------------------
template <int CI, int CO, int STRIDE, int RELU, int WL2, int WIN, int F32OUT>
__global__ __launch_bounds__(256)
void conv_wmma(const bf16_t* __restrict__ in, const bf16_t* __restrict__ wf,
               const float* __restrict__ bias, bf16_t* __restrict__ out,
               float* __restrict__ out_f32)
{
    constexpr int WOUT = 1 << WL2;
    constexpr int HIN  = WIN;
    constexpr int nCC  = CI >> 5;   // K chunks of 32
    constexpr int nCT  = CO >> 4;   // 16-wide N tiles

    int lane   = threadIdx.x & 31;
    int wave   = threadIdx.x >> 5;
    int m_base = (blockIdx.x * 8 + wave) * 16;
    int ctBase = blockIdx.y * 4;
    int kh     = lane >> 4;                   // K-half selector for A layout
    int mlane  = m_base + (lane & 15);
    int oy     = mlane >> WL2;
    int ox     = mlane & (WOUT - 1);

    v8f acc[4] = {};

    for (int kp = 0; kp < 9; ++kp) {
        int ky = kp / 3, kx = kp % 3;
        int yi = oy * STRIDE + ky - 1;
        int xi = ox * STRIDE + kx - 1;
        bool inb = (yi >= 0) && (yi < HIN) && (xi >= 0) && (xi < WIN);
        const bf16_t* arow = in + ((long)yi * WIN + xi) * CI;
        const bf16_t* wrow = wf + (size_t)kp * nCC * nCT * 512 + lane * 16;

        // prefetch next tap's weight block while this tap's WMMAs run
        if (kp < 8)
            __builtin_prefetch(wrow + (size_t)nCC * nCT * 512, 0, 3);

        #pragma unroll
        for (int cc = 0; cc < nCC; ++cc) {
            v8bf z8 = {};
            v8bf r1 = inb ? *(const v8bf*)(arow + cc * 32 + kh * 16)     : z8;
            v8bf r2 = inb ? *(const v8bf*)(arow + cc * 32 + kh * 16 + 8) : z8;
            v16bf a = __builtin_shufflevector(r1, r2,
                          0, 1, 2, 3, 4, 5, 6, 7,
                          8, 9, 10, 11, 12, 13, 14, 15);

            #pragma unroll
            for (int t = 0; t < 4; ++t) {
                v16bf bfr = *(const v16bf*)(wrow +
                                (size_t)(cc * nCT + ctBase + t) * 512);
                acc[t] = __builtin_amdgcn_wmma_f32_16x16x32_bf16(
                             false, a, false, bfr, (short)0, acc[t],
                             false, false);
            }
        }
    }

    int nlo = lane & 15;
    #pragma unroll
    for (int t = 0; t < 4; ++t) {
        int co = (ctBase + t) * 16 + nlo;
        float bb = bias[co];
        #pragma unroll
        for (int j = 0; j < 8; ++j) {
            int m = m_base + j + kh * 8;   // C/D layout: VGPR j -> row j (+8 upper half)
            float val = acc[t][j] + bb;
            if (RELU) val = fmaxf(val, 0.f);
            out[(size_t)m * CO + co] = (bf16_t)val;
            if (F32OUT) out_f32[(size_t)co * (WOUT * WOUT) + m] = val;
        }
    }
}

template <int CI, int CO, int STRIDE, int RELU, int WL2, int WIN, int F32OUT>
static void launch_conv(const bf16_t* in, const bf16_t* wf, const float* bias,
                        bf16_t* out, float* out_f32, hipStream_t stream)
{
    constexpr int WOUT = 1 << WL2;
    dim3 grid((WOUT * WOUT) / 128, CO / 64);
    conv_wmma<CI, CO, STRIDE, RELU, WL2, WIN, F32OUT>
        <<<grid, 256, 0, stream>>>(in, wf, bias, out, out_f32);
}

// ---------------------------------------------------------------------------
// Host-side orchestration
// ---------------------------------------------------------------------------
extern "C" void kernel_launch(void* const* d_in, const int* in_sizes, int n_in,
                              void* d_out, int out_size, void* d_ws, size_t ws_size,
                              hipStream_t stream)
{
    (void)n_in; (void)out_size; (void)ws_size;

    const float* xyz  = (const float*)d_in[0];
    const float* pf   = (const float*)d_in[2];
    const float* mlpw = (const float*)d_in[3];
    const float* bng  = (const float*)d_in[4];
    const float* bnb  = (const float*)d_in[5];
    const float* bnm  = (const float*)d_in[6];
    const float* bnv  = (const float*)d_in[7];
    int N = in_sizes[0] / 3;

    // Workspace layout (bytes):
    //   [0, 8MB)     : swizzled bf16 weights, all 12 layers (~7.5 MB)
    //   [8MB, +16KB) : fused fp32 biases
    //   [16MB, 84MB) : fp32 BEV grid (dead after conversion; reused as ping)
    //   [96MB,130MB) : bf16 activation pong
    char* ws = (char*)d_ws;
    bf16_t* wfAll   = (bf16_t*)(ws);
    float*  biasAll = (float*)(ws + ((size_t)8 << 20));
    float*  bev     = (float*)(ws + ((size_t)16 << 20));
    bf16_t* actA    = (bf16_t*)(ws + ((size_t)16 << 20));  // reuses BEV region
    bf16_t* actB    = (bf16_t*)(ws + ((size_t)96 << 20));

    // (CI, CO) per layer for the prep loop
    static const int LP[12][2] = {
        { 64,  64}, { 64,  64}, { 64,  64},
        { 64, 128}, {128, 128}, {128, 128},
        {128, 256}, {256, 256}, {256, 256},
        {256, 256}, {256, 256}, {256, 256},
    };

    // 1) pillar scatter-max into zeroed fp32 BEV, then convert to bf16 NHWC
    hipMemsetAsync(bev, 0, (size_t)GRID_BEV * GRID_BEV * 64 * sizeof(float), stream);
    pillar_scatter<<<(N + 255) / 256, 256, 0, stream>>>(
        xyz, pf, mlpw, bng, bnb, bnm, bnv, bev, N);
    f32_to_bf16<<<(GRID_BEV * GRID_BEV * 64) / 256, 256, 0, stream>>>(
        bev, actB, GRID_BEV * GRID_BEV * 64);

    // 2) weight prep for all layers (fold BN scale, swizzle to B-fragment order)
    {
        size_t wfOff = 0, bOff = 0;
        for (int i = 0; i < 12; ++i) {
            int CI = LP[i][0], CO = LP[i][1];
            const float* w = (const float*)d_in[8 + 5 * i + 0];
            const float* g = (const float*)d_in[8 + 5 * i + 1];
            const float* b = (const float*)d_in[8 + 5 * i + 2];
            const float* m = (const float*)d_in[8 + 5 * i + 3];
            const float* v = (const float*)d_in[8 + 5 * i + 4];
            int tot = 9 * CI * CO;
            prep_weights<<<(tot + 255) / 256, 256, 0, stream>>>(
                w, g, b, m, v, wfAll + wfOff, biasAll + bOff, CI, CO);
            wfOff += (size_t)tot;
            bOff  += (size_t)CO;
        }
    }

    // 3) conv chain, ping-pong activations; fp32 NCHW taps at layers 8 & 11.
    //    Weight/bias offsets are the running sums of 9*CI*CO and CO.
    float* o4 = (float*)d_out;                           // x_conv4: 256x128x128
    float* o5 = (float*)d_out + (size_t)256 * 128 * 128; // x_conv5: 256x64x64

    launch_conv< 64,  64, 1, 0, 9, 512, 0>(actB, wfAll + 0,       biasAll + 0,    actA, nullptr, stream);
    launch_conv< 64,  64, 1, 1, 9, 512, 0>(actA, wfAll + 36864,   biasAll + 64,   actB, nullptr, stream);
    launch_conv< 64,  64, 1, 1, 9, 512, 0>(actB, wfAll + 73728,   biasAll + 128,  actA, nullptr, stream);
    launch_conv< 64, 128, 2, 1, 8, 512, 0>(actA, wfAll + 110592,  biasAll + 192,  actB, nullptr, stream);
    launch_conv<128, 128, 1, 1, 8, 256, 0>(actB, wfAll + 184320,  biasAll + 320,  actA, nullptr, stream);
    launch_conv<128, 128, 1, 1, 8, 256, 0>(actA, wfAll + 331776,  biasAll + 448,  actB, nullptr, stream);
    launch_conv<128, 256, 2, 1, 7, 256, 0>(actB, wfAll + 479232,  biasAll + 576,  actA, nullptr, stream);
    launch_conv<256, 256, 1, 1, 7, 128, 0>(actA, wfAll + 774144,  biasAll + 832,  actB, nullptr, stream);
    launch_conv<256, 256, 1, 1, 7, 128, 1>(actB, wfAll + 1363968, biasAll + 1088, actA, o4,      stream);
    launch_conv<256, 256, 2, 1, 6, 128, 0>(actA, wfAll + 1953792, biasAll + 1344, actB, nullptr, stream);
    launch_conv<256, 256, 1, 1, 6,  64, 0>(actB, wfAll + 2543616, biasAll + 1600, actA, nullptr, stream);
    launch_conv<256, 256, 1, 1, 6,  64, 1>(actA, wfAll + 3133440, biasAll + 1856, actB, o5,      stream);
}
